// GroupGemmOp_38792144618090
// MI455X (gfx1250) — matrix operand order repaired and checked by reading, not compiled, not run on hardware
//
#include <hip/hip_runtime.h>

typedef __attribute__((ext_vector_type(16))) __bf16       v16bf;
typedef __attribute__((ext_vector_type(8)))  float        v8f;
typedef __attribute__((ext_vector_type(8)))  unsigned int v8u;

#define GK 2048
#define GN 8192
#define BM 128
#define BN 128
#define BK 32
#define LDT 40   // padded K-stride in bf16 elems: 80B rows -> 16B aligned, conflict-free frag reads

__device__ __forceinline__ unsigned short f2bf(float x) {
  unsigned int u = __float_as_uint(x);
  u += 0x7FFFu + ((u >> 16) & 1u);   // round-to-nearest-even
  return (unsigned short)(u >> 16);
}
__device__ __forceinline__ float bf2f(unsigned short h) {
  return __uint_as_float(((unsigned int)h) << 16);
}
__device__ __forceinline__ void split2(float x, float y, unsigned int &hi, unsigned int &lo) {
  unsigned short hx = f2bf(x), hy = f2bf(y);
  unsigned short lx = f2bf(x - bf2f(hx));
  unsigned short ly = f2bf(y - bf2f(hy));
  hi = (unsigned int)hx | ((unsigned int)hy << 16);
  lo = (unsigned int)lx | ((unsigned int)ly << 16);
}

// ---------------- pre-pass 1: A[M][K] f32 -> hi/lo bf16 planes [M][K] ----------------
__global__ __launch_bounds__(256) void split_rows(const float* __restrict__ X,
                                                  unsigned short* __restrict__ hi,
                                                  unsigned short* __restrict__ lo) {
  const size_t i4 = (size_t)blockIdx.x * 256 + threadIdx.x;   // one float4 per thread
  const float4 v = ((const float4*)X)[i4];
  unsigned int h0, l0, h1, l1;
  split2(v.x, v.y, h0, l0);
  split2(v.z, v.w, h1, l1);
  ((uint2*)hi)[i4] = make_uint2(h0, h1);
  ((uint2*)lo)[i4] = make_uint2(l0, l1);
}

// ------- pre-pass 2: B[K][N] f32 -> hi/lo bf16 planes TRANSPOSED [N][K] (LDS-tiled) -------
__global__ __launch_bounds__(256) void split_transpose(const float* __restrict__ B,
                                                       unsigned short* __restrict__ thi,
                                                       unsigned short* __restrict__ tlo) {
  __shared__ __attribute__((aligned(16))) unsigned short sh[64 * LDT];
  __shared__ __attribute__((aligned(16))) unsigned short sl[64 * LDT];
  const int t  = threadIdx.x;
  const int n0 = blockIdx.x * 64;
  const int k0 = blockIdx.y * 32;
#pragma unroll
  for (int i = 0; i < 2; ++i) {                 // 32x64 f32 tile, 512 float4
    int idx = t + i * 256;
    int kr  = idx >> 4;                         // 16 float4 per 64-col row
    int nc  = (idx & 15) << 2;
    const float4 v = *(const float4*)(B + (size_t)(k0 + kr) * GN + n0 + nc);
    float f[4] = {v.x, v.y, v.z, v.w};
#pragma unroll
    for (int j = 0; j < 4; ++j) {
      unsigned short h = f2bf(f[j]);
      unsigned short l = f2bf(f[j] - bf2f(h));
      sh[(nc + j) * LDT + kr] = h;
      sl[(nc + j) * LDT + kr] = l;
    }
  }
  __syncthreads();
  const int r = t >> 2, seg = t & 3;            // 64 rows x 4 segs of 16B
  const uint4 vh = *(const uint4*)&sh[r * LDT + seg * 8];
  const uint4 vl = *(const uint4*)&sl[r * LDT + seg * 8];
  *(uint4*)(thi + (size_t)(n0 + r) * GK + k0 + seg * 8) = vh;
  *(uint4*)(tlo + (size_t)(n0 + r) * GK + k0 + seg * 8) = vl;
}

// ---------------- fast GEMM: preconverted bf16 hi/lo, async LDS staging ----------------
__global__ __launch_bounds__(256, 1) void ggemm_pre(
    const unsigned short* __restrict__ Ahi, const unsigned short* __restrict__ Alo,
    const unsigned short* __restrict__ Bthi, const unsigned short* __restrict__ Btlo,
    float* __restrict__ C) {
  __shared__ __attribute__((aligned(16))) unsigned short sAhi[BM * LDT];
  __shared__ __attribute__((aligned(16))) unsigned short sAlo[BM * LDT];
  __shared__ __attribute__((aligned(16))) unsigned short sBhi[BN * LDT];
  __shared__ __attribute__((aligned(16))) unsigned short sBlo[BN * LDT];

  const int tid   = threadIdx.x;
  const int lane  = tid & 31;
  const int half  = lane >> 4;
  const int l16   = lane & 15;
  const int wid   = tid >> 5;
  const int waveM = wid & 1;    // 2 waves along M (64 rows each)
  const int waveN = wid >> 1;   // 4 waves along N (32 cols each)

  const int n0 = blockIdx.x * BN;
  const int m0 = blockIdx.y * BM;

  // staging assignment: thread -> (row, 32B segment); each async b128 copies 16B
  const int srow = tid >> 1;
  const int sseg = tid & 1;
  const unsigned short* pAh = Ahi  + (size_t)(m0 + srow) * GK + sseg * 16;
  const unsigned short* pAl = Alo  + (size_t)(m0 + srow) * GK + sseg * 16;
  const unsigned short* pBh = Bthi + (size_t)(n0 + srow) * GK + sseg * 16;
  const unsigned short* pBl = Btlo + (size_t)(n0 + srow) * GK + sseg * 16;
  const unsigned dAh = (unsigned)(uintptr_t)&sAhi[srow * LDT + sseg * 16];
  const unsigned dAl = (unsigned)(uintptr_t)&sAlo[srow * LDT + sseg * 16];
  const unsigned dBh = (unsigned)(uintptr_t)&sBhi[srow * LDT + sseg * 16];
  const unsigned dBl = (unsigned)(uintptr_t)&sBlo[srow * LDT + sseg * 16];

  // one stage = 8 async b128 copies (offset: applies to BOTH global and LDS address)
#define STAGE()                                                                        \
  do {                                                                                 \
    asm volatile("global_load_async_to_lds_b128 %0, %1, off\n\t"                       \
                 "global_load_async_to_lds_b128 %0, %1, off offset:16"                 \
                 :: "v"(dAh), "v"(pAh) : "memory");                                    \
    asm volatile("global_load_async_to_lds_b128 %0, %1, off\n\t"                       \
                 "global_load_async_to_lds_b128 %0, %1, off offset:16"                 \
                 :: "v"(dAl), "v"(pAl) : "memory");                                    \
    asm volatile("global_load_async_to_lds_b128 %0, %1, off\n\t"                       \
                 "global_load_async_to_lds_b128 %0, %1, off offset:16"                 \
                 :: "v"(dBh), "v"(pBh) : "memory");                                    \
    asm volatile("global_load_async_to_lds_b128 %0, %1, off\n\t"                       \
                 "global_load_async_to_lds_b128 %0, %1, off offset:16"                 \
                 :: "v"(dBl), "v"(pBl) : "memory");                                    \
    pAh += BK; pAl += BK; pBh += BK; pBl += BK;                                        \
  } while (0)

  v8f acc[4][2];
#pragma unroll
  for (int mt = 0; mt < 4; ++mt)
#pragma unroll
    for (int nt = 0; nt < 2; ++nt) acc[mt][nt] = {};

  STAGE();  // prologue
  for (int k0 = 0; k0 < GK; k0 += BK) {
    asm volatile("s_wait_asynccnt 0x0" ::: "memory");
    __syncthreads();

    // ---- fragments per ISA 7.12.2 layouts ----
    v16bf ahi[4], alo[4], bhi[2], blo[2];
#pragma unroll
    for (int mt = 0; mt < 4; ++mt) {
      const int r = waveM * 64 + mt * 16 + l16;
      const unsigned short* ph = &sAhi[r * LDT];
      const unsigned short* pl = &sAlo[r * LDT];
      v8u uh, ul;
#pragma unroll
      for (int i = 0; i < 8; ++i) {
        const int k = (i < 4 ? 2 * i : 16 + 2 * (i - 4)) + 8 * half;  // A 16x32 map
        uh[i] = *(const unsigned int*)(ph + k);
        ul[i] = *(const unsigned int*)(pl + k);
      }
      ahi[mt] = __builtin_bit_cast(v16bf, uh);
      alo[mt] = __builtin_bit_cast(v16bf, ul);
    }
#pragma unroll
    for (int nt = 0; nt < 2; ++nt) {
      const int n = waveN * 32 + nt * 16 + l16;
      const unsigned short* ph = &sBhi[n * LDT];
      const unsigned short* pl = &sBlo[n * LDT];
      v8u uh, ul;
#pragma unroll
      for (int i = 0; i < 8; ++i) {
        const int k = (i < 4 ? 2 * i : 16 + 2 * (i - 4)) + 8 * half;  // B 32x16 map
        uh[i] = *(const unsigned int*)(ph + k);
        ul[i] = *(const unsigned int*)(pl + k);
      }
      bhi[nt] = __builtin_bit_cast(v16bf, uh);
      blo[nt] = __builtin_bit_cast(v16bf, ul);
    }

    // all LDS reads landed -> safe to let next stage overwrite after barrier
    asm volatile("s_wait_dscnt 0x0" ::: "memory");
    __syncthreads();
    if (k0 + BK < GK) STAGE();   // async DMA overlaps the WMMAs below

    // ---- bf16x3: hi*hi + hi*lo + lo*hi chained into the same f32 acc ----
#pragma unroll
    for (int mt = 0; mt < 4; ++mt)
#pragma unroll
      for (int nt = 0; nt < 2; ++nt) {
        v8f c = acc[mt][nt];
        c = __builtin_amdgcn_wmma_f32_16x16x32_bf16(false, ahi[mt], false, bhi[nt], (short)0, c, false, false);
        c = __builtin_amdgcn_wmma_f32_16x16x32_bf16(false, ahi[mt], false, blo[nt], (short)0, c, false, false);
        c = __builtin_amdgcn_wmma_f32_16x16x32_bf16(false, alo[mt], false, bhi[nt], (short)0, c, false, false);
        acc[mt][nt] = c;
      }
  }
#undef STAGE

  // ---- write back: C/D layout m = r + 8*half, n = lane&15 ----
#pragma unroll
  for (int mt = 0; mt < 4; ++mt) {
    const int gmBase = m0 + waveM * 64 + mt * 16 + 8 * half;
#pragma unroll
    for (int nt = 0; nt < 2; ++nt) {
      const int gn = n0 + waveN * 32 + nt * 16 + l16;
#pragma unroll
      for (int r = 0; r < 8; ++r) {
        C[(size_t)(gmBase + r) * GN + gn] = acc[mt][nt][r];
      }
    }
  }
}

// ---------------- fallback: all-in-one kernel (in-loop conversion) ----------------
__global__ __launch_bounds__(256, 1) void ggemm_bf16x3(
    const float* __restrict__ A, const float* __restrict__ B,
    float* __restrict__ C) {
  __shared__ __attribute__((aligned(16))) unsigned short sAhi[BM * LDT];
  __shared__ __attribute__((aligned(16))) unsigned short sAlo[BM * LDT];
  __shared__ __attribute__((aligned(16))) unsigned short sBhi[BN * LDT];
  __shared__ __attribute__((aligned(16))) unsigned short sBlo[BN * LDT];

  const int tid   = threadIdx.x;
  const int lane  = tid & 31;
  const int half  = lane >> 4;
  const int l16   = lane & 15;
  const int wid   = tid >> 5;
  const int waveM = wid & 1;
  const int waveN = wid >> 1;

  const int n0 = blockIdx.x * BN;
  const int m0 = blockIdx.y * BM;

  v8f acc[4][2];
#pragma unroll
  for (int mt = 0; mt < 4; ++mt)
#pragma unroll
    for (int nt = 0; nt < 2; ++nt) acc[mt][nt] = {};

  for (int k0 = 0; k0 < GK; k0 += BK) {
#pragma unroll
    for (int i = 0; i < 4; ++i) {
      int idx = tid + i * 256;
      int row = idx >> 3;
      int c4  = (idx & 7) << 2;
      const float4 v = *(const float4*)(A + (size_t)(m0 + row) * GK + k0 + c4);
      unsigned int h0, l0, h1, l1;
      split2(v.x, v.y, h0, l0);
      split2(v.z, v.w, h1, l1);
      unsigned short* ph = &sAhi[row * LDT + c4];
      unsigned short* pl = &sAlo[row * LDT + c4];
      *(unsigned int*)(ph)     = h0;
      *(unsigned int*)(ph + 2) = h1;
      *(unsigned int*)(pl)     = l0;
      *(unsigned int*)(pl + 2) = l1;
    }
#pragma unroll
    for (int i = 0; i < 4; ++i) {
      int idx = tid + i * 256;
      int kr  = idx >> 5;
      int nb  = (idx & 31) << 2;
      const float4 v = *(const float4*)(B + (size_t)(k0 + kr) * GN + n0 + nb);
      float f[4] = {v.x, v.y, v.z, v.w};
#pragma unroll
      for (int j = 0; j < 4; ++j) {
        unsigned short h = f2bf(f[j]);
        unsigned short l = f2bf(f[j] - bf2f(h));
        sBhi[(nb + j) * LDT + kr] = h;
        sBlo[(nb + j) * LDT + kr] = l;
      }
    }
    __syncthreads();

    v16bf ahi[4], alo[4], bhi[2], blo[2];
#pragma unroll
    for (int mt = 0; mt < 4; ++mt) {
      const int r = waveM * 64 + mt * 16 + l16;
      const unsigned short* ph = &sAhi[r * LDT];
      const unsigned short* pl = &sAlo[r * LDT];
      v8u uh, ul;
#pragma unroll
      for (int i = 0; i < 8; ++i) {
        const int k = (i < 4 ? 2 * i : 16 + 2 * (i - 4)) + 8 * half;
        uh[i] = *(const unsigned int*)(ph + k);
        ul[i] = *(const unsigned int*)(pl + k);
      }
      ahi[mt] = __builtin_bit_cast(v16bf, uh);
      alo[mt] = __builtin_bit_cast(v16bf, ul);
    }
#pragma unroll
    for (int nt = 0; nt < 2; ++nt) {
      const int n = waveN * 32 + nt * 16 + l16;
      const unsigned short* ph = &sBhi[n * LDT];
      const unsigned short* pl = &sBlo[n * LDT];
      v8u uh, ul;
#pragma unroll
      for (int i = 0; i < 8; ++i) {
        const int k = (i < 4 ? 2 * i : 16 + 2 * (i - 4)) + 8 * half;
        uh[i] = *(const unsigned int*)(ph + k);
        ul[i] = *(const unsigned int*)(pl + k);
      }
      bhi[nt] = __builtin_bit_cast(v16bf, uh);
      blo[nt] = __builtin_bit_cast(v16bf, ul);
    }

#pragma unroll
    for (int mt = 0; mt < 4; ++mt)
#pragma unroll
      for (int nt = 0; nt < 2; ++nt) {
        v8f c = acc[mt][nt];
        c = __builtin_amdgcn_wmma_f32_16x16x32_bf16(false, ahi[mt], false, bhi[nt], (short)0, c, false, false);
        c = __builtin_amdgcn_wmma_f32_16x16x32_bf16(false, ahi[mt], false, blo[nt], (short)0, c, false, false);
        c = __builtin_amdgcn_wmma_f32_16x16x32_bf16(false, alo[mt], false, bhi[nt], (short)0, c, false, false);
        acc[mt][nt] = c;
      }
    __syncthreads();
  }

#pragma unroll
  for (int mt = 0; mt < 4; ++mt) {
    const int gmBase = m0 + waveM * 64 + mt * 16 + 8 * half;
#pragma unroll
    for (int nt = 0; nt < 2; ++nt) {
      const int gn = n0 + waveN * 32 + nt * 16 + l16;
#pragma unroll
      for (int r = 0; r < 8; ++r) {
        C[(size_t)(gmBase + r) * GN + gn] = acc[mt][nt][r];
      }
    }
  }
}

extern "C" void kernel_launch(void* const* d_in, const int* in_sizes, int n_in,
                              void* d_out, int out_size, void* d_ws, size_t ws_size,
                              hipStream_t stream) {
  (void)in_sizes; (void)n_in; (void)out_size;
  const int MS[8] = {4096, 1024, 2048, 3072, 512, 1536, 2560, 768};
  float* out = (float*)d_out;
  const size_t need = 4ull * ((size_t)16384 * GK + 8ull * (size_t)GN * GK);  // hi+lo planes, 2B ea

  if (d_ws != nullptr && ws_size >= need) {
    unsigned short* w = (unsigned short*)d_ws;
    unsigned short *ahi[8], *alo[8], *bhi[8], *blo[8];
    size_t p = 0;
    for (int g = 0; g < 8; ++g) {
      ahi[g] = w + p; p += (size_t)MS[g] * GK;
      alo[g] = w + p; p += (size_t)MS[g] * GK;
      bhi[g] = w + p; p += (size_t)GN * GK;
      blo[g] = w + p; p += (size_t)GN * GK;
    }
    for (int g = 0; g < 8; ++g) {
      const float* A = (const float*)d_in[2 * g];
      const float* B = (const float*)d_in[2 * g + 1];
      split_rows<<<dim3((unsigned)((size_t)MS[g] * GK / 1024)), dim3(256), 0, stream>>>(A, ahi[g], alo[g]);
      split_transpose<<<dim3(GN / 64, GK / 32), dim3(256), 0, stream>>>(B, bhi[g], blo[g]);
    }
    size_t off = 0;
    for (int g = 0; g < 8; ++g) {
      ggemm_pre<<<dim3(GN / BN, MS[g] / BM), dim3(256), 0, stream>>>(
          ahi[g], alo[g], bhi[g], blo[g], out + off);
      off += (size_t)MS[g] * GN;
    }
  } else {
    size_t off = 0;
    for (int g = 0; g < 8; ++g) {
      const float* A = (const float*)d_in[2 * g];
      const float* B = (const float*)d_in[2 * g + 1];
      ggemm_bf16x3<<<dim3(GN / BN, MS[g] / BM), dim3(256), 0, stream>>>(A, B, out + off);
      off += (size_t)MS[g] * GN;
    }
  }
}
